// WindowAttentionRefactored_62130996904477
// MI455X (gfx1250) — matrix operand order repaired and checked by reading, not compiled, not run on hardware
//
#include <hip/hip_runtime.h>

typedef __attribute__((ext_vector_type(16))) _Float16 v16h;
typedef __attribute__((ext_vector_type(8)))  float    v8f;

#define T_   8
#define H_   256
#define W_   256
#define C_   128
#define WSZ  8
#define NTOK 64
#define NH_  4
#define HD_  32

// LDS strides (in _Float16 elements) with padding to dodge bank conflicts
#define XS 136   // 128-wide rows (x window / pre-proj out)
#define HS 40    // 32-wide rows (q/k/v per head)
#define AS 72    // 64-wide rows (attn probs per head)

// LDS byte offsets
#define OFF_X  0
#define OFF_Q  17408
#define OFF_K  37888
#define OFF_V  58368
#define OFF_A  78848
#define LDS_BYTES 115712

// workspace byte offsets
#define WSO_QKVW 0        // 384*128 f16
#define WSO_PRJW 98304    // 128*128 f16
#define WSO_BIAS 131072   // 4*64*64 f32

// ---- WMMA fragment loaders (contiguous k = 16*half + s, wave32) ----

// A fragment 16x32 f16: lane (L%16)=row M, half=L/16 selects K groups
// v0..7: K = half*16 + {0..15}
__device__ __forceinline__ v16h load_a_frag(const _Float16* S, int ld, int lane) {
    const int half = lane >> 4, m = lane & 15;
    const _Float16* row = S + m * ld;
    v16h a;
#pragma unroll
    for (int v = 0; v < 8; ++v) {
        int kk = (half << 4) + (v << 1);
        a[2 * v]     = row[kk];
        a[2 * v + 1] = row[kk + 1];
    }
    return a;
}

// B fragment 32x16 f16 where source is stored N-major (row n, col k):
// B[k][n] = S[n*ld + k].  lanes 0-15: K=0..15 (2/VGPR), lanes 16-31: K=16..31
__device__ __forceinline__ v16h load_b_t(const _Float16* S, int ld, int lane) {
    const int half = lane >> 4, n = lane & 15;
    const _Float16* row = S + n * ld;
    v16h b;
#pragma unroll
    for (int v = 0; v < 8; ++v) {
        int kk = (half << 4) + (v << 1);
        b[2 * v]     = row[kk];
        b[2 * v + 1] = row[kk + 1];
    }
    return b;
}

// B fragment 32x16 f16 where source is stored K-major: B[k][n] = S[k*ld + n]
__device__ __forceinline__ v16h load_b_d(const _Float16* S, int ld, int lane) {
    const int half = lane >> 4, n = lane & 15;
    v16h b;
#pragma unroll
    for (int v = 0; v < 8; ++v) {
        int kk = (half << 4) + (v << 1);
        b[2 * v]     = S[kk * ld + n];
        b[2 * v + 1] = S[(kk + 1) * ld + n];
    }
    return b;
}

__device__ __forceinline__ v8f wmma_f16(v16h a, v16h b, v8f c) {
    return __builtin_amdgcn_wmma_f32_16x16x32_f16(false, a, false, b, (short)0, c,
                                                  false, false);
}

// ---- prep: f32->f16 weights, dense relative-position bias table ----
__global__ void prep_kernel(const float* __restrict__ qkv_w,
                            const float* __restrict__ proj_w,
                            const float* __restrict__ rpb,
                            const int*   __restrict__ rpi,
                            _Float16* __restrict__ qkv_wh,
                            _Float16* __restrict__ proj_wh,
                            float* __restrict__ biasp) {
    int tid = blockIdx.x * 256 + threadIdx.x;           // 49152 threads
    qkv_wh[tid] = (_Float16)qkv_w[tid];
    if (tid < 128 * 128) proj_wh[tid] = (_Float16)proj_w[tid];
    if (tid < NH_ * NTOK * NTOK) {                      // bias[h][m][n]
        int h = tid >> 12, mn = tid & 4095;
        biasp[tid] = rpb[rpi[mn] * NH_ + h];
    }
}

// ---- fused window attention: 1 block = 1 window, 1 wave = 1 head ----
__launch_bounds__(128)
__global__ void winattn_kernel(const float* __restrict__ vid,
                               const float* __restrict__ qkv_b,
                               const float* __restrict__ proj_b,
                               const _Float16* __restrict__ qkv_wh,
                               const _Float16* __restrict__ proj_wh,
                               const float* __restrict__ biasp,
                               float* __restrict__ out) {
    extern __shared__ char smem[];
    _Float16* xh = (_Float16*)(smem + OFF_X);                       // 64 x XS (reused as out)
    const int tid  = threadIdx.x;
    const int lane = tid & 31;
    const int wid  = tid >> 5;                                      // head id
    _Float16* qh = (_Float16*)(smem + OFF_Q) + wid * NTOK * HS;
    _Float16* kh = (_Float16*)(smem + OFF_K) + wid * NTOK * HS;
    _Float16* vh = (_Float16*)(smem + OFF_V) + wid * NTOK * HS;
    _Float16* ah = (_Float16*)(smem + OFF_A) + wid * NTOK * AS;

    const int b  = blockIdx.x;
    const int t  = b >> 10;
    const int wh = (b >> 5) & 31;
    const int ww = b & 31;
    const float* xin  = vid + (((size_t)t * H_ + wh * WSZ) * W_ + ww * WSZ) * C_;
    float*       outw = out + (((size_t)t * H_ + wh * WSZ) * W_ + ww * WSZ) * C_;

    // ---- stage 1a: window -> LDS (f16) ----
#pragma unroll
    for (int it = 0; it < 16; ++it) {
        int idx = tid + it * 128;          // 2048 float4 per window
        int tok = idx >> 5, c4 = idx & 31;
        int i = tok >> 3, j = tok & 7;
        const float4 f = *(const float4*)(xin + ((size_t)i * W_ + j) * C_ + c4 * 4);
        _Float16* dst = xh + tok * XS + c4 * 4;
        dst[0] = (_Float16)f.x; dst[1] = (_Float16)f.y;
        dst[2] = (_Float16)f.z; dst[3] = (_Float16)f.w;
    }
    __syncthreads();

    const int n    = lane & 15;
    const int half = lane >> 4;
    const float scale = 0.17677669529663687f;   // 1/sqrt(32)

    // ---- stage 1b: q,k,v for this head (q pre-scaled) ----
#pragma unroll
    for (int g = 0; g < 3; ++g) {
        const int dbase = g * C_ + wid * HD_;
        _Float16* dst = (g == 0) ? qh : (g == 1) ? kh : vh;
        const float smul = (g == 0) ? scale : 1.0f;
#pragma unroll
        for (int mt = 0; mt < 4; ++mt) {
#pragma unroll
            for (int nt = 0; nt < 2; ++nt) {
                v8f acc = {};
#pragma unroll
                for (int ks = 0; ks < 4; ++ks) {
                    v16h a  = load_a_frag(xh + mt * 16 * XS + ks * 32, XS, lane);
                    v16h bw = load_b_t(qkv_wh + (size_t)(dbase + nt * 16) * C_ + ks * 32,
                                       C_, lane);
                    acc = wmma_f16(a, bw, acc);
                }
                const float bv = qkv_b[dbase + nt * 16 + n];
#pragma unroll
                for (int vi = 0; vi < 8; ++vi) {
                    int m = mt * 16 + vi + 8 * half;
                    dst[m * HS + nt * 16 + n] = (_Float16)((acc[vi] + bv) * smul);
                }
            }
        }
    }
    __syncthreads();   // xh now free for reuse as pre-projection output

    // ---- stage 2: attn = softmax(q k^T + bias), probs -> LDS f16 ----
    v16h aq[4];
#pragma unroll
    for (int mt = 0; mt < 4; ++mt) aq[mt] = load_a_frag(qh + mt * 16 * HS, HS, lane);

#pragma unroll
    for (int mt = 0; mt < 4; ++mt) {
        v8f acc[4];
#pragma unroll
        for (int nt = 0; nt < 4; ++nt) {
            // seed accumulator with relative-position bias (D layout)
            const float* bp = biasp + ((size_t)(wid * 64 + mt * 16 + 8 * half) * 64)
                              + nt * 16 + n;
            v8f c;
#pragma unroll
            for (int vi = 0; vi < 8; ++vi) c[vi] = bp[(size_t)vi * 64];
            v16h bk = load_b_t(kh + (nt * 16) * HS, HS, lane);   // B[k][n]=K[n][k]
            acc[nt] = wmma_f16(aq[mt], bk, c);
        }
        // softmax per row: row = mt*16 + vi + 8*half lives across 16 lanes of a half
#pragma unroll
        for (int vi = 0; vi < 8; ++vi) {
            float x0 = acc[0][vi], x1 = acc[1][vi], x2 = acc[2][vi], x3 = acc[3][vi];
            float mx = fmaxf(fmaxf(x0, x1), fmaxf(x2, x3));
#pragma unroll
            for (int msk = 8; msk >= 1; msk >>= 1)
                mx = fmaxf(mx, __shfl_xor(mx, msk, 32));   // masks<16: stays in half
            float e0 = __expf(x0 - mx), e1 = __expf(x1 - mx);
            float e2 = __expf(x2 - mx), e3 = __expf(x3 - mx);
            float s = e0 + e1 + e2 + e3;
#pragma unroll
            for (int msk = 8; msk >= 1; msk >>= 1)
                s += __shfl_xor(s, msk, 32);
            float inv = 1.0f / s;
            int m = mt * 16 + vi + 8 * half;
            ah[m * AS +  0 + n] = (_Float16)(e0 * inv);
            ah[m * AS + 16 + n] = (_Float16)(e1 * inv);
            ah[m * AS + 32 + n] = (_Float16)(e2 * inv);
            ah[m * AS + 48 + n] = (_Float16)(e3 * inv);
        }
    }

    // ---- stage 2b: out_h = P @ V -> xh[:, wid*32 .. wid*32+31] ----
#pragma unroll
    for (int mt = 0; mt < 4; ++mt) {
#pragma unroll
        for (int nt = 0; nt < 2; ++nt) {
            v8f acc = {};
#pragma unroll
            for (int ks = 0; ks < 2; ++ks) {
                v16h a  = load_a_frag(ah + mt * 16 * AS + ks * 32, AS, lane);
                v16h bv = load_b_d(vh + (ks * 32) * HS + nt * 16, HS, lane);
                acc = wmma_f16(a, bv, acc);
            }
#pragma unroll
            for (int vi = 0; vi < 8; ++vi) {
                int m = mt * 16 + vi + 8 * half;
                xh[m * XS + wid * HD_ + nt * 16 + n] = (_Float16)acc[vi];
            }
        }
    }
    __syncthreads();

    // ---- stage 3: output projection, all 4 waves split 32 tiles ----
    for (int idx = wid; idx < 32; idx += 4) {
        int mt = idx >> 3, nt = idx & 7;
        v8f acc = {};
#pragma unroll
        for (int ks = 0; ks < 4; ++ks) {
            v16h a  = load_a_frag(xh + mt * 16 * XS + ks * 32, XS, lane);
            v16h bw = load_b_t(proj_wh + (size_t)(nt * 16) * C_ + ks * 32, C_, lane);
            acc = wmma_f16(a, bw, acc);
        }
        const float pb = proj_b[nt * 16 + n];
#pragma unroll
        for (int vi = 0; vi < 8; ++vi) {
            int m = mt * 16 + vi + 8 * half;
            int i = m >> 3, j = m & 7;
            outw[((size_t)i * W_ + j) * C_ + nt * 16 + n] = acc[vi] + pb;
        }
    }
}

extern "C" void kernel_launch(void* const* d_in, const int* in_sizes, int n_in,
                              void* d_out, int out_size, void* d_ws, size_t ws_size,
                              hipStream_t stream) {
    const float* vid    = (const float*)d_in[0];
    const float* qkv_w  = (const float*)d_in[1];
    const float* qkv_b  = (const float*)d_in[2];
    const float* proj_w = (const float*)d_in[3];
    const float* proj_b = (const float*)d_in[4];
    const float* rpb    = (const float*)d_in[5];
    const int*   rpi    = (const int*)d_in[6];

    _Float16* qkv_wh  = (_Float16*)((char*)d_ws + WSO_QKVW);
    _Float16* proj_wh = (_Float16*)((char*)d_ws + WSO_PRJW);
    float*    biasp   = (float*)((char*)d_ws + WSO_BIAS);

    prep_kernel<<<192, 256, 0, stream>>>(qkv_w, proj_w, rpb, rpi,
                                         qkv_wh, proj_wh, biasp);

    const int nwin = T_ * (H_ / WSZ) * (W_ / WSZ);   // 8192
    winattn_kernel<<<nwin, 128, LDS_BYTES, stream>>>(vid, qkv_b, proj_b,
                                                     qkv_wh, proj_wh, biasp,
                                                     (float*)d_out);
}
